// SoftHardMultiheadAttention_43980465111374
// MI455X (gfx1250) — matrix operand order repaired and checked by reading, not compiled, not run on hardware
//
#include <hip/hip_runtime.h>

typedef __attribute__((ext_vector_type(16))) __bf16 v16bf;
typedef __attribute__((ext_vector_type(8)))  __bf16 v8bf;
typedef __attribute__((ext_vector_type(4)))  __bf16 v4bf;
typedef __attribute__((ext_vector_type(8)))  float  v8f;

#define NH    16
#define HD    64
#define TLEN  1024
#define SLEN  1024
#define BSZ   4
#define ED    1024
#define ROWS  (TLEN * BSZ)   // 4096 rows of the projection GEMMs
#define SCP   1028           // padded f32 stride for score/avg LDS (bank de-conflict)

// ---------------------------------------------------------------------------
// WMMA helpers (CDNA5 v_wmma_f32_16x16x32_bf16, wave32)
// ---------------------------------------------------------------------------
__device__ __forceinline__ v8f wmma_bf16(v16bf a, v16bf b, v8f c) {
  return __builtin_amdgcn_wmma_f32_16x16x32_bf16(
      false, a, false, b, (short)0, c, false, false);
}

// A-matrix 16x32 bf16 fragment from row-major (M x K) tile, leading dim ldk.
// Lane l: M = l%16 ; K = (e/8)*16 + (l/16)*8 + (e%8)
__device__ __forceinline__ v16bf load_frag_a(const __bf16* base, int ldk) {
  int l  = threadIdx.x & 31;
  int m  = l & 15;
  int kh = (l >> 4) << 3;
  union { v16bf v; v8bf h[2]; } u;
  u.h[0] = *(const v8bf*)(base + m * ldk + kh);
  u.h[1] = *(const v8bf*)(base + m * ldk + kh + 16);
  return u.v;
}

// Same fragment sourced from f32 data (per-element convert, packs to bf16).
__device__ __forceinline__ v16bf load_frag_a_f32(const float* base, int ldk) {
  int l  = threadIdx.x & 31;
  int m  = l & 15;
  int kh = (l >> 4) << 3;
  const float* p0 = base + m * ldk + kh;
  const float* p1 = p0 + 16;
  v16bf r;
#pragma unroll
  for (int i = 0; i < 8; ++i) {
    r[i]     = (__bf16)p0[i];
    r[i + 8] = (__bf16)p1[i];
  }
  return r;
}

// B-matrix 32x16 bf16 fragment from [N][K]-stored data, leading dim ldk.
// Lane l: N = l%16 ; K = (e/8)*16 + (l/16)*8 + (e%8)  (2x16B)
__device__ __forceinline__ v16bf load_frag_b(const __bf16* base, int ldk) {
  int l  = threadIdx.x & 31;
  int n  = l & 15;
  int k0 = (l >> 4) << 3;
  union { v16bf v; v8bf h[2]; } u;
  u.h[0] = *(const v8bf*)(base + n * ldk + k0);
  u.h[1] = *(const v8bf*)(base + n * ldk + k0 + 16);
  return u.v;
}

// Stage a [TROWS x 32] f32 tile into LDS as bf16. Compile-time trip count
// (TROWS*32/512 iterations, uniform across lanes) -> fully unrolled,
// float4 global loads + packed bf16x4 b64 LDS stores, no EXEC juggling.
template <int TROWS>
__device__ __forceinline__ void stage_tile(const float* __restrict__ src,
                                           size_t ld, __bf16* __restrict__ dst,
                                           int tid) {
#pragma unroll
  for (int it = 0; it < (TROWS * 32) / 512; ++it) {
    int i = it * 512 + tid * 4;
    int r = i >> 5, k = i & 31;
    const float4 xv = *(const float4*)(src + (size_t)r * ld + k);
    v4bf o;
    o[0] = (__bf16)xv.x; o[1] = (__bf16)xv.y;
    o[2] = (__bf16)xv.z; o[3] = (__bf16)xv.w;
    *(v4bf*)(dst + r * 32 + k) = o;
  }
}

// ---------------------------------------------------------------------------
// GEMM core: block tile 128 rows x 64 cols, K step 32, 4 waves.
// Wave w owns rows [w*32, w*32+32): 2 A-frags x 4 B-frags -> 8 wmma / K-step.
// ---------------------------------------------------------------------------
struct Acc { v8f a[2][4]; };

__device__ __forceinline__ void gemm_core(const float* __restrict__ X, size_t ldx,
                                          const float* __restrict__ W, size_t ldw,
                                          int row0, int c0, int tid,
                                          __bf16 (*lds_x)[32], __bf16 (*lds_w)[32],
                                          Acc& acc) {
  const int w = tid >> 5;
#pragma unroll
  for (int s = 0; s < 2; ++s)
#pragma unroll
    for (int nt = 0; nt < 4; ++nt) acc.a[s][nt] = v8f{};

  for (int kk = 0; kk < ED; kk += 32) {
    stage_tile<128>(X + (size_t)row0 * ldx + kk, ldx, &lds_x[0][0], tid);
    stage_tile<64>(W + (size_t)c0 * ldw + kk, ldw, &lds_w[0][0], tid);
    __syncthreads();

    v16bf a0 = load_frag_a(&lds_x[w * 32][0], 32);
    v16bf a1 = load_frag_a(&lds_x[w * 32 + 16][0], 32);
#pragma unroll
    for (int nt = 0; nt < 4; ++nt) {
      v16bf b = load_frag_b(&lds_w[nt * 16][0], 32);
      acc.a[0][nt] = wmma_bf16(a0, b, acc.a[0][nt]);
      acc.a[1][nt] = wmma_bf16(a1, b, acc.a[1][nt]);
    }
    __syncthreads();
  }
}

// ---------------------------------------------------------------------------
// Kernel 1: QKV projection.  C(4096 x 3072) = X(4096 x 1024) @ W^T + bias
// X = query for cols < 1024, key otherwise. Scatter bf16 into workspace:
// Q[b][h][t][d] (scaled 1/8), K[b][h][s][d], V^T[b][h][d][s].
// Column-region branch is BLOCK-UNIFORM (64 | 1024) -> scalar branch only.
// ---------------------------------------------------------------------------
__global__ void gemm_qkv_kernel(const float* __restrict__ query,
                                const float* __restrict__ key,
                                const float* __restrict__ wproj,
                                const float* __restrict__ bias,
                                __bf16* __restrict__ qs,
                                __bf16* __restrict__ ks,
                                __bf16* __restrict__ vt) {
  __shared__ __bf16 lds_x[128][32];
  __shared__ __bf16 lds_w[64][32];

  const int tid  = threadIdx.x;
  const int lane = tid & 31;
  const int w    = tid >> 5;
  const int c0   = blockIdx.x * 64;    // 0..3008 (uniform region)
  const int row0 = blockIdx.y * 128;

  const float* Xg = (c0 < ED) ? query : key;

  Acc acc;
  gemm_core(Xg, ED, wproj, ED, row0, c0, tid, lds_x, lds_w, acc);

  const int n  = lane & 15;
  const int mb = (lane >> 4) << 3;

  if (c0 < ED) {                       // ---- Q region (scaled) ----
#pragma unroll
    for (int s = 0; s < 2; ++s)
#pragma unroll
      for (int nt = 0; nt < 4; ++nt)
#pragma unroll
        for (int v = 0; v < 8; ++v) {
          int row = row0 + w * 32 + s * 16 + mb + v;
          int col = c0 + nt * 16 + n;
          float val = (acc.a[s][nt][v] + bias[col]) * 0.125f;
          int t = row >> 2, b = row & 3;
          int hh = col >> 6, d = col & 63;
          qs[((((size_t)b * NH + hh) * TLEN) + t) * HD + d] = (__bf16)val;
        }
  } else if (c0 < 2 * ED) {            // ---- K region ----
#pragma unroll
    for (int s = 0; s < 2; ++s)
#pragma unroll
      for (int nt = 0; nt < 4; ++nt)
#pragma unroll
        for (int v = 0; v < 8; ++v) {
          int row = row0 + w * 32 + s * 16 + mb + v;
          int col = c0 + nt * 16 + n;
          float val = acc.a[s][nt][v] + bias[col];
          int t = row >> 2, b = row & 3;
          int f = col - ED;
          int hh = f >> 6, d = f & 63;
          ks[((((size_t)b * NH + hh) * SLEN) + t) * HD + d] = (__bf16)val;
        }
  } else {                             // ---- V region (store transposed) ----
#pragma unroll
    for (int s = 0; s < 2; ++s)
#pragma unroll
      for (int nt = 0; nt < 4; ++nt)
#pragma unroll
        for (int v = 0; v < 8; ++v) {
          int row = row0 + w * 32 + s * 16 + mb + v;
          int col = c0 + nt * 16 + n;
          float val = acc.a[s][nt][v] + bias[col];
          int t = row >> 2, b = row & 3;
          int f = col - 2 * ED;
          int hh = f >> 6, d = f & 63;
          vt[((((size_t)b * NH + hh) * HD) + d) * SLEN + t] = (__bf16)val;
        }
  }
}

// ---------------------------------------------------------------------------
// Kernel 2: attention. Block = (b, 16-row t-tile), 128 threads (4 waves).
// ---------------------------------------------------------------------------
__global__ void attention_kernel(const __bf16* __restrict__ qs,
                                 const __bf16* __restrict__ ks,
                                 const __bf16* __restrict__ vt,
                                 const float* __restrict__ hard,
                                 float* __restrict__ og,
                                 float* __restrict__ avg_out) {
  __shared__ float sc[16][SCP];     // scores -> attn (padded stride)
  __shared__ float avg[16][SCP];    // head-sum of attn
  __shared__ float red[16][8];
  __shared__ float rowmax[16];
  __shared__ float rowinv[16];

  const int tid  = threadIdx.x;
  const int lane = tid & 31;
  const int w    = tid >> 5;
  const int t0   = blockIdx.x * 16;
  const int b    = blockIdx.y;
  const int row  = tid >> 3;        // 0..15
  const int part = tid & 7;         // 0..7
  const int n    = lane & 15;
  const int mbv  = (lane >> 4) << 3;

  for (int i = tid; i < 16 * SLEN; i += 128) {
    int r = i >> 10, c = i & 1023;
    avg[r][c] = 0.0f;
  }
  __syncthreads();

  for (int h = 0; h < NH; ++h) {
    // ---- scores = Q Kt : wave w handles s-tiles [w*16, w*16+16) ----
    const __bf16* qbase = qs + (((size_t)b * NH + h) * TLEN + t0) * HD;
    v16bf a0 = load_frag_a(qbase + 0, HD);
    v16bf a1 = load_frag_a(qbase + 32, HD);
    const __bf16* kbase = ks + (((size_t)b * NH + h) * SLEN) * HD;
    for (int st = w * 16; st < w * 16 + 16; ++st) {
      const __bf16* kb = kbase + (size_t)st * 16 * HD;
      __builtin_prefetch(kb + 16 * HD, 0, 1);   // next s-tile -> global_prefetch_b8
      v16bf b0 = load_frag_b(kb + 0, HD);
      v16bf b1 = load_frag_b(kb + 32, HD);
      v8f c = {};
      c = wmma_bf16(a0, b0, c);
      c = wmma_bf16(a1, b1, c);
#pragma unroll
      for (int v = 0; v < 8; ++v) sc[mbv + v][st * 16 + n] = c[v];
    }
    __syncthreads();

    // ---- softmax stats (bank-interleaved row scans) ----
    float mx = -3.4e38f;
    for (int j = 0; j < 128; ++j) mx = fmaxf(mx, sc[row][part + 8 * j]);
    red[row][part] = mx;
    __syncthreads();
    if (part == 0) {
      float m = red[row][0];
#pragma unroll
      for (int j = 1; j < 8; ++j) m = fmaxf(m, red[row][j]);
      rowmax[row] = m;
    }
    __syncthreads();
    float m = rowmax[row];
    float s = 0.0f;
    for (int j = 0; j < 128; ++j) s += __expf(sc[row][part + 8 * j] - m);
    red[row][part] = s;
    __syncthreads();
    if (part == 0) {
      float t = 0.0f;
#pragma unroll
      for (int j = 0; j < 8; ++j) t += red[row][j];
      rowinv[row] = 1.0f / t;
    }
    __syncthreads();

    // ---- attn = softmax * hard_mask ; accumulate avg ----
    const float* mrow = hard + ((size_t)b * TLEN + (t0 + row)) * SLEN;
    float inv = rowinv[row];
    m = rowmax[row];
    for (int j = 0; j < 128; ++j) {
      int c = part + 8 * j;
      float p = __expf(sc[row][c] - m) * inv * mrow[c];
      sc[row][c] = p;
      avg[row][c] += p;
    }
    __syncthreads();

    // ---- out = attn @ V : wave w owns d-tile w ----
    const __bf16* vbase = vt + (((size_t)b * NH + h) * HD + w * 16) * SLEN;
    v8f o = {};
    for (int sk = 0; sk < SLEN / 32; ++sk) {
      __builtin_prefetch(vbase + (sk + 1) * 32, 0, 1);
      v16bf av = load_frag_a_f32(&sc[0][sk * 32], SCP);
      v16bf bv = load_frag_b(vbase + sk * 32, SLEN);
      o = wmma_bf16(av, bv, o);
    }
#pragma unroll
    for (int v = 0; v < 8; ++v) {
      int t = t0 + mbv + v;
      og[((size_t)t * BSZ + b) * ED + h * HD + w * 16 + n] = o[v];
    }
    __syncthreads();   // sc/avg reused next head
  }

  // ---- avg_weights = sum_h(attn) / NH ----
  for (int i = tid; i < 16 * SLEN; i += 128) {
    int r = i >> 10, c = i & 1023;
    avg_out[(((size_t)b * TLEN) + t0 + r) * SLEN + c] = avg[r][c] * (1.0f / NH);
  }
}

// ---------------------------------------------------------------------------
// Kernel 3: output projection. out(4096 x 1024) = O @ Wo^T + bias (f32 out).
// ---------------------------------------------------------------------------
__global__ void gemm_out_kernel(const float* __restrict__ og,
                                const float* __restrict__ wo,
                                const float* __restrict__ bo,
                                float* __restrict__ out) {
  __shared__ __bf16 lds_x[128][32];
  __shared__ __bf16 lds_w[64][32];

  const int tid  = threadIdx.x;
  const int lane = tid & 31;
  const int w    = tid >> 5;
  const int c0   = blockIdx.x * 64;
  const int row0 = blockIdx.y * 128;

  Acc acc;
  gemm_core(og, ED, wo, ED, row0, c0, tid, lds_x, lds_w, acc);

  const int n  = lane & 15;
  const int mb = (lane >> 4) << 3;
#pragma unroll
  for (int s = 0; s < 2; ++s)
#pragma unroll
    for (int nt = 0; nt < 4; ++nt)
#pragma unroll
      for (int v = 0; v < 8; ++v) {
        int row = row0 + w * 32 + s * 16 + mb + v;
        int col = c0 + nt * 16 + n;
        out[(size_t)row * ED + col] = acc.a[s][nt][v] + bo[col];
      }
}

// ---------------------------------------------------------------------------
// Launch
// ---------------------------------------------------------------------------
extern "C" void kernel_launch(void* const* d_in, const int* in_sizes, int n_in,
                              void* d_out, int out_size, void* d_ws, size_t ws_size,
                              hipStream_t stream) {
  (void)in_sizes; (void)n_in; (void)out_size; (void)ws_size;
  const float* query = (const float*)d_in[0];
  const float* key   = (const float*)d_in[1];
  const float* hard  = (const float*)d_in[2];
  const float* wi    = (const float*)d_in[3];
  const float* bi    = (const float*)d_in[4];
  const float* wo    = (const float*)d_in[5];
  const float* bo    = (const float*)d_in[6];

  float* out     = (float*)d_out;                       // (T,B,E)
  float* avg_out = out + (size_t)TLEN * BSZ * ED;       // (B,T,S)

  char* ws = (char*)d_ws;
  __bf16* qs = (__bf16*)(ws);                            //  8 MB
  __bf16* ks = (__bf16*)(ws + (size_t)8  * 1024 * 1024); //  8 MB
  __bf16* vt = (__bf16*)(ws + (size_t)16 * 1024 * 1024); //  8 MB
  float*  og = (float*)(ws + (size_t)24 * 1024 * 1024);  // 16 MB

  dim3 g1(3 * ED / 64, ROWS / 128);
  gemm_qkv_kernel<<<g1, 128, 0, stream>>>(query, key, wi, bi, qs, ks, vt);

  dim3 g2(TLEN / 16, BSZ);
  attention_kernel<<<g2, 128, 0, stream>>>(qs, ks, vt, hard, og, avg_out);

  dim3 g3(ED / 64, ROWS / 128);
  gemm_out_kernel<<<g3, 128, 0, stream>>>(og, wo, bo, out);
}
